// OptEinsumScaledDotProductAttention_70720931495946
// MI455X (gfx1250) — matrix operand order repaired and checked by reading, not compiled, not run on hardware
//
#include <hip/hip_runtime.h>
#include <hip/hip_bf16.h>
#include <cstdint>

#define B_   8
#define H_   8
#define LQ_  1024
#define LK_  1024
#define D_   64
#define MROWS 32          // query rows per workgroup in the attention kernel

#define AS1 __attribute__((address_space(1)))   // global
#define AS3 __attribute__((address_space(3)))   // LDS

typedef __attribute__((ext_vector_type(16))) __bf16 v16bf;
typedef __attribute__((ext_vector_type(8)))  __bf16 v8bf;
typedef __attribute__((ext_vector_type(4)))  __bf16 v4bf;
typedef __attribute__((ext_vector_type(8)))  float  v8f;
typedef __attribute__((ext_vector_type(4)))  float  v4f;
typedef __attribute__((ext_vector_type(4)))  int    v4i;

// ------------------------------------------------------------- async copy ---
// CDNA5 async global->LDS path (ASYNCcnt); falls back to a synchronous 16B
// vector copy if the toolchain doesn't declare the builtins.
#if __has_builtin(__builtin_amdgcn_global_load_async_to_lds_b128) && \
    __has_builtin(__builtin_amdgcn_s_wait_asynccnt)
#define USE_ASYNC_LDS 1
#else
#define USE_ASYNC_LDS 0
#endif

__device__ __forceinline__ void copy16(__bf16* dstLds, const __bf16* srcGlob) {
#if USE_ASYNC_LDS
  __builtin_amdgcn_global_load_async_to_lds_b128(
      (AS1 v4i*)srcGlob, (AS3 v4i*)dstLds, 0, 0);
#else
  *(AS3 v8bf*)dstLds = *(const v8bf*)srcGlob;
#endif
}

template <int N>
__device__ __forceinline__ void async_wait() {
#if USE_ASYNC_LDS
  __builtin_amdgcn_s_wait_asynccnt(N);
#endif
}

// ---------------------------------------------------------------- helpers ---

__device__ __forceinline__ __bf16 f2bf(float x) {
  unsigned u = __builtin_bit_cast(unsigned, x);
  unsigned r = (u + 0x7fffu + ((u >> 16) & 1u)) >> 16;   // round-to-nearest-even
  unsigned short h = (unsigned short)r;
  return __builtin_bit_cast(__bf16, h);
}

__device__ __forceinline__ v8f wmma_bf16(v16bf a, v16bf b, v8f c) {
  return __builtin_amdgcn_wmma_f32_16x16x32_bf16(
      false, a, false, b, (short)0, c, false, false);
}

// A fragment: 16x32 bf16 tile in LDS, row-major, leading dim ldm.
// Lane L: row m = L&15; K runs {8*(L>>4)..+7} and {16+8*(L>>4)..+7} -> 2x ds_load_b128.
__device__ __forceinline__ v16bf load_frag_a(const AS3 __bf16* base, int ldm, int lane) {
  const AS3 __bf16* p = base + (lane & 15) * ldm + ((lane >> 4) << 3);
  v8bf lo = *(const AS3 v8bf*)(p);
  v8bf hi = *(const AS3 v8bf*)(p + 16);
  v16bf r;
#pragma unroll
  for (int t = 0; t < 8; ++t) { r[t] = lo[t]; r[8 + t] = hi[t]; }
  return r;
}

// B fragment from a K-contiguous (transposed) tile in LDS: layout [n][ldk],
// lane L (col n = L&15): K runs {8*(L>>4)..+7} and {16+8*(L>>4)..+7} -> 2x ds_load_b128.
__device__ __forceinline__ v16bf load_frag_bt(const AS3 __bf16* base, int ldk, int lane) {
  const AS3 __bf16* p = base + (lane & 15) * ldk + ((lane >> 4) << 3);
  v8bf lo = *(const AS3 v8bf*)(p);
  v8bf hi = *(const AS3 v8bf*)(p + 16);
  v16bf r;
#pragma unroll
  for (int t = 0; t < 8; ++t) { r[t] = lo[t]; r[8 + t] = hi[t]; }
  return r;
}

// C/D tile (16x16 f32) scatter into LDS: VGPR r -> M = r + 8*(L>=16), N = L&15.
__device__ __forceinline__ void store_frag_c(AS3 float* base, int ldn, int lane, v8f c) {
  AS3 float* p = base + ((lane >> 4) * 8) * ldn + (lane & 15);
#pragma unroll
  for (int rr = 0; rr < 8; ++rr) p[rr * ldn] = c[rr];
}

// --------------------------------------------- kernel 1: fold weight chains --
__global__ __launch_bounds__(256) void fold_w_kernel(
    const float* __restrict__ WA,  const float* __restrict__ WB,
    const float* __restrict__ WAt, const float* __restrict__ WBt,
    const float* __restrict__ Wav, const float* __restrict__ Wbv,
    float* __restrict__ Wout, float* __restrict__ Wvout) {
  __shared__ float T1[4096];
  __shared__ float T2[4096];
  const int b   = blockIdx.x;
  const int tid = threadIdx.x;
  const float* A  = WA  + b * 4096;
  const float* Bm = WB  + b * 4096;
  const float* At = WAt + b * 4096;
  const float* Bt = WBt + b * 4096;
  const float* Av = Wav + b * 4096;
  const float* Bv = Wbv + b * 4096;

  for (int i = tid; i < 4096; i += 256) {
    int r = i >> 6, c = i & 63;
    float s = 0.f;
    for (int k = 0; k < 64; ++k) s += A[r * 64 + k] * Bm[k * 64 + c];
    T1[i] = s;
  }
  __syncthreads();
  for (int i = tid; i < 4096; i += 256) {
    int r = i >> 6, c = i & 63;
    float s = 0.f;
    for (int k = 0; k < 64; ++k) s += T1[r * 64 + k] * Bt[k * 64 + c];
    T2[i] = s;
  }
  __syncthreads();
  for (int i = tid; i < 4096; i += 256) {
    int r = i >> 6, c = i & 63;
    float s = 0.f;
    for (int k = 0; k < 64; ++k) s += T2[r * 64 + k] * At[k * 64 + c];
    Wout[b * 4096 + i] = s;
    float sv = 0.f;
    for (int k = 0; k < 64; ++k) sv += Av[r * 64 + k] * Bv[k * 64 + c];
    Wvout[b * 4096 + i] = sv;
  }
}

// ------------------------------- kernel 2: qW = q @ W[b] -> bf16 [hb][a][j] --
__global__ __launch_bounds__(256) void proj_q_kernel(
    const float* __restrict__ q, const float* __restrict__ W,
    __bf16* __restrict__ qw) {
  __shared__ float Ws[64 * 64];
  __shared__ float Rw[4 * 64];
  const int blk = blockIdx.x;            // 64 heads * 256 row-groups
  const int hb  = blk >> 8;
  const int a0  = (blk & 255) * 4;
  const int k   = hb >> 3, b = hb & 7;
  const int tid = threadIdx.x;
  const float* Wb = W + b * 4096;
  for (int i = tid; i < 4096; i += 256) Ws[i] = Wb[i];
  {
    int rr = tid >> 6, c = tid & 63;
    Rw[tid] = q[(((size_t)k * LQ_ + a0 + rr) * H_ + b) * D_ + c];
  }
  __syncthreads();
  const int rr = tid >> 6, j = tid & 63;
  float s = 0.f;
  for (int c = 0; c < 64; ++c) s += Rw[rr * 64 + c] * Ws[c * 64 + j];
  qw[((size_t)hb * LQ_ + a0 + rr) * D_ + j] = f2bf(s);
}

// ------------------- kernel 3: qt [hb][j][n] f32 -> bf16 transposed [hb][n][j]
__global__ __launch_bounds__(256) void transpose_qt_kernel(
    const float* __restrict__ qt, __bf16* __restrict__ out) {
  __shared__ __bf16 T[64 * 65];          // padded to dodge bank conflicts
  const int hb  = blockIdx.y;
  const int n0  = blockIdx.x * 64;
  const int tid = threadIdx.x;
  const float* src = qt + (size_t)hb * D_ * LK_;
  for (int i = tid; i < 4096; i += 256) {
    int j = i >> 6, n = i & 63;
    T[n * 65 + j] = f2bf(src[(size_t)j * LK_ + n0 + n]);  // coalesced read
  }
  __syncthreads();
  for (int i = tid; i < 4096; i += 256) {
    int n = i >> 6, j = i & 63;
    out[((size_t)hb * LK_ + n0 + n) * D_ + j] = T[n * 65 + j];  // coalesced write
  }
}

// ------------- kernel 4: Vw = v @ Wv[b] -> bf16 transposed layout [hb][c][m] --
__global__ __launch_bounds__(256) void proj_v_kernel(
    const float* __restrict__ v, const float* __restrict__ Wv,
    __bf16* __restrict__ vw) {
  __shared__ float  Ws[4096];
  __shared__ float  Vr[4096];
  __shared__ __bf16 Ob[4096];
  const int hb  = blockIdx.y;
  const int m0  = blockIdx.x * 64;
  const int k   = hb >> 3, b = hb & 7;
  const int tid = threadIdx.x;
  const float* Wb = Wv + b * 4096;
  for (int i = tid; i < 4096; i += 256) Ws[i] = Wb[i];
  for (int i = tid; i < 4096; i += 256) {
    int mm = i >> 6, n = i & 63;
    Vr[i] = v[(((size_t)k * LK_ + m0 + mm) * H_ + b) * D_ + n];
  }
  __syncthreads();
  for (int i = tid; i < 4096; i += 256) {
    int mm = i >> 6, c = i & 63;
    float s = 0.f;
    for (int n = 0; n < 64; ++n) s += Vr[mm * 64 + n] * Ws[n * 64 + c];
    Ob[c * 64 + mm] = f2bf(s);           // transpose in LDS
  }
  __syncthreads();
  for (int i = tid; i < 4096; i += 256)  // coalesced [c][m] write
    vw[((size_t)hb * D_ + (i >> 6)) * LK_ + m0 + (i & 63)] = Ob[i];
}

// ---------------- kernel 5: fused scores -> softmax -> attn write -> output --
// qw:  [64][1024][64] bf16 (A, row-major)
// ktT: [64][1024][64] bf16 (B of scores, K-contiguous per column n)
// vwT: [64][64][1024] bf16 (B of output, K(=m)-contiguous per column c)
__global__ __launch_bounds__(256) void attn_kernel(
    const __bf16* __restrict__ qw, const __bf16* __restrict__ ktT,
    const __bf16* __restrict__ vwT,
    float* __restrict__ outO, float* __restrict__ outA) {
  extern __shared__ char smem[];
  float*  S      = (float*)(smem);               // 32*1024 f32   = 131072 B
  __bf16* P      = (__bf16*)(smem + 131072);     // 32*1024 bf16  =  65536 B
  __bf16* Qs     = (__bf16*)(smem + 196608);     // 32*64  bf16   =   4096 B
  __bf16* Bs0    = (__bf16*)(smem + 200704);     // slab buffer 0 =   8192 B
  __bf16* Bs1    = (__bf16*)(smem + 208896);     // slab buffer 1 =   8192 B
  float*  redp   = (float*)(smem + 217088);      // 256 f32
  float*  rowMax = (float*)(smem + 218112);      // 32 f32
  float*  rowInv = (float*)(smem + 218240);      // 32 f32

  const int hb   = blockIdx.y;
  const int row0 = blockIdx.x * MROWS;
  const int tid  = threadIdx.x;
  const int wave = tid >> 5;                     // wave32: 8 waves
  const int lane = tid & 31;
  __bf16* Bs[2] = {Bs0, Bs1};

  // Stage Q' tile (32x64 = 2048 bf16): 16B per thread
  copy16(Qs + tid * 8,
         qw + (size_t)hb * LQ_ * D_ + (size_t)row0 * D_ + tid * 8);

  // ---- scores: S = Q' @ Kt, double-buffered 64-column slabs ----------------
  const __bf16* ktH = ktT + (size_t)hb * LK_ * D_;     // [n][64], contiguous slabs
  copy16(Bs[0] + tid * 8,        ktH + tid * 8);
  copy16(Bs[0] + 2048 + tid * 8, ktH + 2048 + tid * 8);
  async_wait<0>();
  __syncthreads();

  const int rt  = wave & 1;                      // 16-row tile
  const int ctw = wave >> 1;                     // 16-col tile within slab
  v16bf qa0 = load_frag_a((const AS3 __bf16*)(Qs + rt * 16 * D_),      D_, lane);
  v16bf qa1 = load_frag_a((const AS3 __bf16*)(Qs + rt * 16 * D_ + 32), D_, lane);

  for (int it = 0; it < LK_ / 64; ++it) {
    const int cur = it & 1;
    if (it + 1 < LK_ / 64) {                     // prefetch next slab
      const __bf16* nsrc = ktH + (size_t)(it + 1) * 4096;
      copy16(Bs[cur ^ 1] + tid * 8,        nsrc + tid * 8);
      copy16(Bs[cur ^ 1] + 2048 + tid * 8, nsrc + 2048 + tid * 8);
      async_wait<2>();                           // current slab has landed
    } else {
      async_wait<0>();
    }
    __syncthreads();
    const AS3 __bf16* Bc = (const AS3 __bf16*)(Bs[cur] + (ctw * 16) * 64);
    v16bf b0 = load_frag_bt(Bc,      64, lane);  // j 0..31
    v16bf b1 = load_frag_bt(Bc + 32, 64, lane);  // j 32..63
    v8f acc = {0.f, 0.f, 0.f, 0.f, 0.f, 0.f, 0.f, 0.f};
    acc = wmma_bf16(qa0, b0, acc);
    acc = wmma_bf16(qa1, b1, acc);
    store_frag_c((AS3 float*)(S + rt * 16 * LK_ + it * 64 + ctw * 16),
                 LK_, lane, acc);
    __syncthreads();
  }

  // ---- softmax over LK (f32, scale 1/8), 128-element chunks per thread -----
  const float invT = 0.125f;
  const int r = tid >> 3, sub = tid & 7;
  AS3 float* Srow = (AS3 float*)(S + r * LK_ + sub * 128);
  float mx = -3.0e38f;
  for (int c = 0; c < 128; c += 4) {
    v4f x = *(const AS3 v4f*)(Srow + c);
    mx = fmaxf(mx, fmaxf(fmaxf(x.x, x.y), fmaxf(x.z, x.w)));
  }
  redp[tid] = mx;
  __syncthreads();
  if (tid < 32) {
    float m2 = redp[tid * 8];
#pragma unroll
    for (int t = 1; t < 8; ++t) m2 = fmaxf(m2, redp[tid * 8 + t]);
    rowMax[tid] = m2 * invT;
  }
  __syncthreads();
  const float rm = rowMax[r];
  float sm = 0.f;
  for (int c = 0; c < 128; c += 4) {
    v4f x = *(AS3 v4f*)(Srow + c);
    x.x = __expf(x.x * invT - rm);
    x.y = __expf(x.y * invT - rm);
    x.z = __expf(x.z * invT - rm);
    x.w = __expf(x.w * invT - rm);
    *(AS3 v4f*)(Srow + c) = x;
    sm += (x.x + x.y) + (x.z + x.w);
  }
  redp[tid] = sm;
  __syncthreads();
  if (tid < 32) {
    float s2 = 0.f;
#pragma unroll
    for (int t = 0; t < 8; ++t) s2 += redp[tid * 8 + t];
    rowInv[tid] = 1.f / s2;
  }
  __syncthreads();

  // Normalize; single coalesced f32 write of attn + packed bf16 copy in LDS
  {
    AS1 float* dstA = (AS1 float*)(outA + (size_t)hb * LQ_ * LK_ +
                                   (size_t)row0 * LK_);
    for (int i = tid * 4; i < MROWS * LK_; i += 1024) {
      v4f s4 = *(const AS3 v4f*)((AS3 float*)S + i);
      float ri = rowInv[i >> 10];
      v4f p4 = {s4.x * ri, s4.y * ri, s4.z * ri, s4.w * ri};
      *(AS1 v4f*)(dstA + i) = p4;                      // global_store_b128
      v4bf pb;
      pb[0] = f2bf(p4.x); pb[1] = f2bf(p4.y);
      pb[2] = f2bf(p4.z); pb[3] = f2bf(p4.w);
      *(AS3 v4bf*)((AS3 __bf16*)P + i) = pb;           // ds_store_b64
    }
  }
  __syncthreads();

  // ---- output: O = P(32x1024) @ Vw(1024x64), double-buffered 32-k slabs ----
  const int nt = wave >> 1;                      // output col tile 0..3
  const __bf16* vwH = vwT + (size_t)hb * D_ * LK_;     // [c][1024]
  const int cc = tid >> 2, part = (tid & 3) * 8; // 16B chunk per thread
  copy16(Bs[0] + cc * 32 + part, vwH + (size_t)cc * LK_ + part);
  v8f acc = {0.f, 0.f, 0.f, 0.f, 0.f, 0.f, 0.f, 0.f};
  for (int kb = 0; kb < LK_; kb += 32) {
    const int cur = (kb >> 5) & 1;
    if (kb + 32 < LK_) {
      copy16(Bs[cur ^ 1] + cc * 32 + part,
             vwH + (size_t)cc * LK_ + kb + 32 + part);
      async_wait<1>();
    } else {
      async_wait<0>();
    }
    __syncthreads();
    v16bf a = load_frag_a((const AS3 __bf16*)(P + rt * 16 * LK_ + kb), LK_, lane);
    v16bf b = load_frag_bt((const AS3 __bf16*)(Bs[cur] + (nt * 16) * 32), 32, lane);
    acc = wmma_bf16(a, b, acc);
    __syncthreads();
  }
  {
    AS1 float* dst = (AS1 float*)(outO + (size_t)hb * LQ_ * D_ +
                                  (size_t)(row0 + rt * 16) * D_ + nt * 16);
    int nn = lane & 15, m0 = (lane >> 4) * 8;
#pragma unroll
    for (int rr = 0; rr < 8; ++rr) dst[(m0 + rr) * D_ + nn] = acc[rr];
  }
}

// ----------------------------------------------------------------- launcher --

extern "C" void kernel_launch(void* const* d_in, const int* in_sizes, int n_in,
                              void* d_out, int out_size, void* d_ws, size_t ws_size,
                              hipStream_t stream) {
  (void)in_sizes; (void)n_in; (void)out_size; (void)ws_size;
  const float* q   = (const float*)d_in[0];
  const float* WA  = (const float*)d_in[1];
  const float* WB  = (const float*)d_in[2];
  const float* WAt = (const float*)d_in[3];
  const float* WBt = (const float*)d_in[4];
  const float* Wav = (const float*)d_in[5];
  const float* Wbv = (const float*)d_in[6];
  const float* qt  = (const float*)d_in[7];
  const float* v   = (const float*)d_in[8];

  char* ws = (char*)d_ws;
  float*  Wf  = (float*)ws;                                  // 8*4096 f32
  float*  Wvf = (float*)(ws + 131072);                       // 8*4096 f32
  __bf16* qWb = (__bf16*)(ws + 262144);                      // [64][1024][64]
  __bf16* qtT = (__bf16*)(ws + 262144 + 8388608);            // [64][1024][64]
  __bf16* vWT = (__bf16*)(ws + 262144 + 2 * 8388608);        // [64][64][1024]

  float* outO = (float*)d_out;                               // [64][1024][64]
  float* outA = outO + (size_t)B_ * H_ * LQ_ * D_;           // [64][1024][1024]

  fold_w_kernel<<<8, 256, 0, stream>>>(WA, WB, WAt, WBt, Wav, Wbv, Wf, Wvf);
  proj_q_kernel<<<16384, 256, 0, stream>>>(q, Wf, qWb);
  transpose_qt_kernel<<<dim3(16, 64), 256, 0, stream>>>(qt, qtT);
  proj_v_kernel<<<dim3(16, 64), 256, 0, stream>>>(v, Wvf, vWT);

  const int SMEM = 218368;  // ~213 KB dynamic LDS (WGP has 320 KB)
  (void)hipFuncSetAttribute((const void*)attn_kernel,
                            hipFuncAttributeMaxDynamicSharedMemorySize, SMEM);
  attn_kernel<<<dim3(LQ_ / MROWS, B_ * H_), 256, SMEM, stream>>>(
      qWb, qtT, vWT, outO, outA);
}
